// BiDirectionalMinGRU_42563125903625
// MI455X (gfx1250) — matrix operand order repaired and checked by reading, not compiled, not run on hardware
//
#include <hip/hip_runtime.h>
#include <hip/hip_bf16.h>

// ---------------------------------------------------------------------------
// BiDirectional minGRU for MI455X (gfx1250, wave32, WMMA).
// GEMMs use split-bf16 (hi+lo) with 3x v_wmma_f32_16x16x32_bf16 for ~f32
// precision at bf16 tensor-core rates. Everything else is memory-path tuned.
// Assumes ws_size >= ~800 MB.
// ---------------------------------------------------------------------------

#define kB   32
#define kL   4096
#define kH   256
#define kNR  (kB * kL)          // 131072 rows

typedef __attribute__((ext_vector_type(16))) __bf16 v16bf;
typedef __attribute__((ext_vector_type(8)))  float  v8f;

union Frag16 { v16bf v; uint4 q[2]; };

// ---------------------------------------------------------------------------
// Split an f32 matrix into bf16 hi + bf16 lo (residual).
// ---------------------------------------------------------------------------
__global__ void split_w_kernel(const float* __restrict__ src,
                               __bf16* __restrict__ hi, __bf16* __restrict__ lo,
                               int n) {
  int i = blockIdx.x * blockDim.x + threadIdx.x;
  if (i < n) {
    float v = src[i];
    __bf16 h = (__bf16)v;
    hi[i] = h;
    lo[i] = (__bf16)(v - (float)h);
  }
}

// ---------------------------------------------------------------------------
// Input projection (K=3): inp = x @ W_in + b, written as split bf16.
// One block per (b,l) row, 256 threads = h.
// ---------------------------------------------------------------------------
__global__ void input_proj_kernel(const float* __restrict__ x,
                                  const float* __restrict__ Wf, const float* __restrict__ bf,
                                  const float* __restrict__ Wb, const float* __restrict__ bb,
                                  __bf16* __restrict__ fhi, __bf16* __restrict__ flo,
                                  __bf16* __restrict__ bhi, __bf16* __restrict__ blo) {
  size_t row = blockIdx.x;
  int h = threadIdx.x;
  float x0 = x[row * 3 + 0], x1 = x[row * 3 + 1], x2 = x[row * 3 + 2];
  float sf = fmaf(x0, Wf[h], fmaf(x1, Wf[kH + h], fmaf(x2, Wf[2 * kH + h], bf[h])));
  float sb = fmaf(x0, Wb[h], fmaf(x1, Wb[kH + h], fmaf(x2, Wb[2 * kH + h], bb[h])));
  size_t o = row * kH + h;
  __bf16 sfh = (__bf16)sf;
  fhi[o] = sfh;  flo[o] = (__bf16)(sf - (float)sfh);
  __bf16 sbh = (__bf16)sb;
  bhi[o] = sbh;  blo[o] = (__bf16)(sb - (float)sbh);
}

// ---------------------------------------------------------------------------
// Fused gate GEMM: z = sigmoid(A@Wz + bz), ht = tanh(A@Wh + bh).
// Block = 256 threads = 8 waves. Waves 0-3: z for M-subtiles 0-3.
// Waves 4-7: ht for M-subtiles 0-3. M tile = 64 rows, full N = 256, K = 256
// looped in chunks of 32 (one bf16 WMMA K-step).
//
// WMMA operand layouts (CDNA5 ISA 7.12.2):
//   A 16x32 bf16 : lane -> M (lane&15); regs pack K:
//                  K = 16*(j/4) + 8*(lane>>4) + 2*(j&3) + half
//                  => two contiguous 16B loads at kbase and kbase+16.
//   B 32x16 bf16 : lane -> N (lane&15); regs pack K in the same order as A.
//   C/D 16x16 f32: lane -> N (lane&15); VGPR j -> M = j + 8*(lane>>4).
//
// Weights staged in 64KB LDS per 32-K chunk (4 mats: Wz_hi/lo, Wh_hi/lo),
// XOR-swizzled at 16B granularity so B-frag reads (lane = N, 512B row
// stride) are bank-conflict-free. A frags load straight from global (the
// 2-wave duplication hits WGP$).
// ---------------------------------------------------------------------------
__global__ void __launch_bounds__(256)
gates_gemm_kernel(const __bf16* __restrict__ Ahi, const __bf16* __restrict__ Alo,
                  const __bf16* __restrict__ Wzhi, const __bf16* __restrict__ Wzlo,
                  const __bf16* __restrict__ Whhi, const __bf16* __restrict__ Whlo,
                  const float* __restrict__ bz, const float* __restrict__ bh,
                  float* __restrict__ zout, float* __restrict__ htout) {
  __shared__ uint4 Blds[4 * 32 * 32];   // [mat][k(32)][granule(32 x 16B)] = 64KB

  const int tid  = threadIdx.x;
  const int wave = tid >> 5;
  const int lane = tid & 31;
  const int sub  = wave & 3;            // M subtile within 64-row tile
  const int doH  = wave >> 2;           // 0 => z/Wz/sigmoid, 1 => ht/Wh/tanh
  const int rowBase = blockIdx.x * 64 + sub * 16;
  const int mrow   = rowBase + (lane & 15);
  const int khalf  = (lane >> 4) * 8;

  v8f zero8 = {0.f, 0.f, 0.f, 0.f, 0.f, 0.f, 0.f, 0.f};
  v8f acc[16];
#pragma unroll
  for (int n = 0; n < 16; ++n) acc[n] = zero8;

  const int matHi = doH ? 2 : 0;        // LDS matrix index for this wave's hi
  const int kB_   = lane;               // unused; B frag column is lane & 15

  for (int kc = 0; kc < 8; ++kc) {
    __syncthreads();                    // previous chunk fully consumed
    // Cooperative load: 4 mats x 32 rows x 32 granules = 4096 uint4 / 256 thr.
#pragma unroll
    for (int i = 0; i < 16; ++i) {
      const __bf16* src = (i >> 2) == 0 ? Wzhi
                        : (i >> 2) == 1 ? Wzlo
                        : (i >> 2) == 2 ? Whhi : Whlo;   // constant per unrolled i
      int rem = ((i & 3) << 8) + tid;   // 0..1023 within matrix
      int k = rem >> 5;                 // row 0..31 of chunk
      int g = rem & 31;                 // 16B granule 0..31
      uint4 val = *((const uint4*)(src + (size_t)(kc * 32 + k) * kH) + g);
      Blds[(i >> 2) * 1024 + k * 32 + (g ^ k)] = val;    // swizzled store
    }
    __syncthreads();

    // A fragments straight from global.
    Frag16 ahi, alo;
    {
      const __bf16* ph = Ahi + (size_t)mrow * kH + kc * 32 + khalf;
      const __bf16* pl = Alo + (size_t)mrow * kH + kc * 32 + khalf;
      ahi.q[0] = *(const uint4*)(ph);
      ahi.q[1] = *(const uint4*)(ph + 16);
      alo.q[0] = *(const uint4*)(pl);
      alo.q[1] = *(const uint4*)(pl + 16);
    }

#pragma unroll
    for (int n = 0; n < 16; ++n) {
      Frag16 bhi, blo;
      const __bf16* Lb = (const __bf16*)Blds;
      const int c = n * 16 + (lane & 15);
#pragma unroll
      for (int s = 0; s < 16; ++s) {
        int k = (s < 8) ? (khalf + s) : (16 + khalf + (s - 8));
        int gi = k * 32 + ((c >> 3) ^ k);
        bhi.v[s] = Lb[(matHi * 1024 + gi) * 8 + (c & 7)];
        blo.v[s] = Lb[((matHi + 1) * 1024 + gi) * 8 + (c & 7)];
      }
      // split-bf16 f32 emulation: hi*hi + lo*hi + hi*lo
      acc[n] = __builtin_amdgcn_wmma_f32_16x16x32_bf16(false, ahi.v, false, bhi.v,
                                                       (short)0, acc[n], false, false);
      acc[n] = __builtin_amdgcn_wmma_f32_16x16x32_bf16(false, alo.v, false, bhi.v,
                                                       (short)0, acc[n], false, false);
      acc[n] = __builtin_amdgcn_wmma_f32_16x16x32_bf16(false, ahi.v, false, blo.v,
                                                       (short)0, acc[n], false, false);
    }
  }

  // Epilogue: bias + activation + store (lanes 0-15 / 16-31 each write a
  // contiguous 64B span per (n,j)).
  const float* bias = doH ? bh : bz;
  float* outp = doH ? htout : zout;
  const int ncol  = lane & 15;
  const int rhalf = 8 * (lane >> 4);
#pragma unroll
  for (int n = 0; n < 16; ++n) {
    float bv = bias[n * 16 + ncol];
#pragma unroll
    for (int j = 0; j < 8; ++j) {
      float xv = acc[n][j] + bv;
      float y;
      if (doH) {                         // tanh(x) = (1-e^-2x)/(1+e^-2x)
        float e = __expf(-2.f * xv);
        y = (1.f - e) / (1.f + e);
      } else {                           // sigmoid
        y = 1.f / (1.f + __expf(-xv));
      }
      size_t row = (size_t)rowBase + rhalf + j;
      outp[row * kH + n * 16 + ncol] = y;
    }
  }
}

// ---------------------------------------------------------------------------
// Sequential gated scan, one thread per (b,h) chain. Writes the SHIFTED state
// h_used[l] = state after steps < l (fwd) / steps > l (bwd). Unroll-4 with
// prefetch to hide HBM latency; reads/writes coalesce across h.
// ---------------------------------------------------------------------------
__global__ void scan_gru_kernel(const float* __restrict__ z, const float* __restrict__ ht,
                                float* __restrict__ hout, int forward) {
  int c = blockIdx.x * blockDim.x + threadIdx.x;   // chain = b*kH + h
  int b = c >> 8;
  int h = c & 255;
  size_t base = (size_t)b * kL * kH + h;
  float hp = 0.f;
  for (int l0 = 0; l0 < kL; l0 += 4) {
    float zv[4], hv[4];
#pragma unroll
    for (int i = 0; i < 4; ++i) {
      int l = forward ? (l0 + i) : (kL - 1 - (l0 + i));
      size_t idx = base + (size_t)l * kH;
      zv[i] = z[idx];
      hv[i] = ht[idx];
    }
    if (l0 + 4 < kL) {
      int ln = forward ? (l0 + 4) : (kL - 1 - (l0 + 4));
      __builtin_prefetch(z + base + (size_t)ln * kH, 0, 0);
      __builtin_prefetch(ht + base + (size_t)ln * kH, 0, 0);
    }
#pragma unroll
    for (int i = 0; i < 4; ++i) {
      int l = forward ? (l0 + i) : (kL - 1 - (l0 + i));
      hout[base + (size_t)l * kH] = hp;
      hp = fmaf(zv[i], hv[i] - hp, hp);            // (1-z)h + z*ht
    }
  }
}

// ---------------------------------------------------------------------------
// Output head: out[r] = dot(hf[r,:], Wg[0:H]) + dot(hb[r,:], Wg[H:2H]) + bg.
// wave32 shfl reduce + LDS cross-wave reduce.
// ---------------------------------------------------------------------------
__global__ void out_proj_kernel(const float* __restrict__ hf, const float* __restrict__ hb,
                                const float* __restrict__ Wg, const float* __restrict__ bg,
                                float* __restrict__ out) {
  __shared__ float partial[8];
  size_t r = blockIdx.x;
  int t = threadIdx.x;
  float v = hf[r * kH + t] * Wg[t] + hb[r * kH + t] * Wg[kH + t];
#pragma unroll
  for (int off = 16; off > 0; off >>= 1) v += __shfl_down(v, off, 32);
  if ((t & 31) == 0) partial[t >> 5] = v;
  __syncthreads();
  if (t < 8) {
    float s = partial[t];
#pragma unroll
    for (int off = 4; off > 0; off >>= 1) s += __shfl_down(s, off, 8);
    if (t == 0) out[r] = s + bg[0];
  }
}

// ---------------------------------------------------------------------------
extern "C" void kernel_launch(void* const* d_in, const int* in_sizes, int n_in,
                              void* d_out, int out_size, void* d_ws, size_t ws_size,
                              hipStream_t stream) {
  const float* x     = (const float*)d_in[0];
  const float* Wf_in = (const float*)d_in[1];
  const float* bf_in = (const float*)d_in[2];
  const float* Wb_in = (const float*)d_in[3];
  const float* bb_in = (const float*)d_in[4];
  const float* Wz_f  = (const float*)d_in[5];
  const float* bz_f  = (const float*)d_in[6];
  const float* Wh_f  = (const float*)d_in[7];
  const float* bh_f  = (const float*)d_in[8];
  const float* Wz_b  = (const float*)d_in[9];
  const float* bz_b  = (const float*)d_in[10];
  const float* Wh_b  = (const float*)d_in[11];
  const float* bh_b  = (const float*)d_in[12];
  const float* Wg    = (const float*)d_in[13];
  const float* bg    = (const float*)d_in[14];
  float* out = (float*)d_out;

  const size_t NH = (size_t)kNR * kH;              // 33,554,432 elements
  const int    WN = kH * kH;                       // 65,536 per weight matrix

  // ---- workspace layout (~769 MB) ----
  __bf16* inf_hi = (__bf16*)d_ws;                  //  64MB  \  aliased later by
  __bf16* inf_lo = inf_hi + NH;                    //  64MB   | hf (fwd states)
  __bf16* inb_hi = inf_lo + NH;                    //  64MB   | hb (bwd states)
  __bf16* inb_lo = inb_hi + NH;                    //  64MB  /
  __bf16* wsp    = inb_lo + NH;                    //  8 x 128KB weight splits
  __bf16* wzf_hi = wsp + 0 * WN;  __bf16* wzf_lo = wsp + 1 * WN;
  __bf16* whf_hi = wsp + 2 * WN;  __bf16* whf_lo = wsp + 3 * WN;
  __bf16* wzb_hi = wsp + 4 * WN;  __bf16* wzb_lo = wsp + 5 * WN;
  __bf16* whb_hi = wsp + 6 * WN;  __bf16* whb_lo = wsp + 7 * WN;
  float* z_f  = (float*)(wsp + 8 * WN);            // 128MB each
  float* ht_f = z_f  + NH;
  float* z_b  = ht_f + NH;
  float* ht_b = z_b  + NH;
  float* hf   = (float*)d_ws;                      // alias: inp splits dead
  float* hb   = hf + NH;                           // by the time scan runs

  // 1) weight splits
  split_w_kernel<<<WN / 256, 256, 0, stream>>>(Wz_f, wzf_hi, wzf_lo, WN);
  split_w_kernel<<<WN / 256, 256, 0, stream>>>(Wh_f, whf_hi, whf_lo, WN);
  split_w_kernel<<<WN / 256, 256, 0, stream>>>(Wz_b, wzb_hi, wzb_lo, WN);
  split_w_kernel<<<WN / 256, 256, 0, stream>>>(Wh_b, whb_hi, whb_lo, WN);

  // 2) input projections (both directions), split-bf16 A
  input_proj_kernel<<<kNR, kH, 0, stream>>>(x, Wf_in, bf_in, Wb_in, bb_in,
                                            inf_hi, inf_lo, inb_hi, inb_lo);

  // 3) fused gate GEMMs (WMMA)
  gates_gemm_kernel<<<kNR / 64, 256, 0, stream>>>(inf_hi, inf_lo,
                                                  wzf_hi, wzf_lo, whf_hi, whf_lo,
                                                  bz_f, bh_f, z_f, ht_f);
  gates_gemm_kernel<<<kNR / 64, 256, 0, stream>>>(inb_hi, inb_lo,
                                                  wzb_hi, wzb_lo, whb_hi, whb_lo,
                                                  bz_b, bh_b, z_b, ht_b);

  // 4) sequential scans (fwd + bwd); overwrite A-split region with states
  scan_gru_kernel<<<(kB * kH) / 64, 64, 0, stream>>>(z_f, ht_f, hf, 1);
  scan_gru_kernel<<<(kB * kH) / 64, 64, 0, stream>>>(z_b, ht_b, hb, 0);

  // 5) output head
  out_proj_kernel<<<kNR, kH, 0, stream>>>(hf, hb, Wg, bg, out);
}
